// Transformer_28303834481146
// MI455X (gfx1250) — compile-verified, matrix operand order checked by reading, not run on hardware
//
#include <hip/hip_runtime.h>
#include <hip/hip_bf16.h>
#include <math.h>
#include <stdint.h>

// ---------------------------------------------------------------------------
// Types
// ---------------------------------------------------------------------------
typedef __bf16 bf16_t;
typedef __attribute__((ext_vector_type(16))) __bf16 v16bf;
typedef __attribute__((ext_vector_type(8)))  __bf16 v8bf;
typedef __attribute__((ext_vector_type(8)))  float  v8f;

__device__ __forceinline__ v8f wmma_bf16(v16bf a, v16bf b, v8f c) {
  // (neg_a, A, neg_b, B, c_mod, C, reuse_a, reuse_b)
  return __builtin_amdgcn_wmma_f32_16x16x32_bf16(false, a, false, b, (short)0, c,
                                                 false, false);
}

// bit-exact fp32 -> bf16 (round-to-nearest-even)
__device__ __forceinline__ bf16_t f2bf(float x) {
  unsigned u = __float_as_uint(x);
  unsigned r = (u + 0x7FFFu + ((u >> 16) & 1u)) >> 16;
  unsigned short s = (unsigned short)r;
  bf16_t out;
  __builtin_memcpy(&out, &s, 2);
  return out;
}

// ---------------------------------------------------------------------------
// CDNA5 async global->LDS copy (ASYNCcnt-tracked, bypasses VGPRs).
// ISA 15.18.3: GLOBAL_LOAD_ASYNC_TO_LDS_B128  vdst(lds byte addr), vaddr, off
// Per-lane 16B; one instruction moves 512B per wave.
// ---------------------------------------------------------------------------
__device__ __forceinline__ void async_b128(const bf16_t* g, bf16_t* l) {
  unsigned lds = (unsigned)(uintptr_t)l;  // generic LDS ptr low 32b = LDS offset
  asm volatile("global_load_async_to_lds_b128 %0, %1, off"
               :
               : "v"(lds), "v"(g)
               : "memory");
}
__device__ __forceinline__ void wait_async0() {
  asm volatile("s_wait_asynccnt 0x0" ::: "memory");
}
__device__ __forceinline__ void wait_async4() {
  asm volatile("s_wait_asynccnt 0x4" ::: "memory");
}

// ---------------------------------------------------------------------------
// Fragment loader: ISA 16-bit A-matrix 16x32 layout (B uses same layout on
// [N][K]-stored data).  lane<16: K 0..7 (v0..3), K 16..23 (v4..7);
// lane>=16: K 8..15 / 24..31.  STRIDE in elements, rows 16B aligned.
// ---------------------------------------------------------------------------
template <int STRIDE>
__device__ __forceinline__ v16bf load_frag(const bf16_t* base, int row0, int k0) {
  int lane = threadIdx.x & 31;
  int row  = row0 + (lane & 15);
  int kh   = (lane >> 4) * 8;
  const bf16_t* p = base + row * STRIDE + k0 + kh;
  v8bf lo = *(const v8bf*)(p);
  v8bf hi = *(const v8bf*)(p + 16);
  v16bf r;
#pragma unroll
  for (int i = 0; i < 8; ++i) { r[i] = lo[i]; r[i + 8] = hi[i]; }
  return r;
}

// ---------------------------------------------------------------------------
// Weight conversion: fp32 [K][N] -> bf16 [N][K] (transposed), 32x32 LDS tiles
// ---------------------------------------------------------------------------
__global__ __launch_bounds__(256) void convert_transpose_k(
    const float* __restrict__ in, bf16_t* __restrict__ out, int K, int N) {
  __shared__ float t[32][33];
  int tx = threadIdx.x & 31, ty = threadIdx.x >> 5;  // 32 x 8
  int n0 = blockIdx.x * 32, k0 = blockIdx.y * 32;
#pragma unroll
  for (int i = 0; i < 4; ++i)
    t[ty + 8 * i][tx] = in[(size_t)(k0 + ty + 8 * i) * N + n0 + tx];
  __syncthreads();
#pragma unroll
  for (int i = 0; i < 4; ++i)
    out[(size_t)(n0 + ty + 8 * i) * K + k0 + tx] = f2bf(t[tx][ty + 8 * i]);
}

// ---------------------------------------------------------------------------
// LayerNorm: one block per token row, D = 1024
// ---------------------------------------------------------------------------
template <bool OUT_F32>
__global__ __launch_bounds__(256) void layernorm_k(
    const float* __restrict__ x, const float* __restrict__ sc,
    const float* __restrict__ bi, void* __restrict__ out, int D) {
  int row = blockIdx.x;
  const float* xr = x + (size_t)row * D;
  float ls = 0.f, lq = 0.f;
  for (int i = threadIdx.x; i < D; i += 256) {
    float v = xr[i];
    ls += v;
    lq += v * v;
  }
#pragma unroll
  for (int m = 16; m >= 1; m >>= 1) {
    ls += __shfl_xor(ls, m, 32);
    lq += __shfl_xor(lq, m, 32);
  }
  __shared__ float red[2][8];
  int wave = threadIdx.x >> 5;
  if ((threadIdx.x & 31) == 0) { red[0][wave] = ls; red[1][wave] = lq; }
  __syncthreads();
  float ts = 0.f, tq = 0.f;
#pragma unroll
  for (int w = 0; w < 8; ++w) { ts += red[0][w]; tq += red[1][w]; }
  float mu   = ts / (float)D;
  float var  = tq / (float)D - mu * mu;
  float rinv = rsqrtf(var + 1e-5f);
  for (int i = threadIdx.x; i < D; i += 256) {
    float v = (xr[i] - mu) * rinv * sc[i] + bi[i];
    if (OUT_F32)
      ((float*)out)[(size_t)row * D + i] = v;
    else
      ((bf16_t*)out)[(size_t)row * D + i] = f2bf(v);
  }
}

// ---------------------------------------------------------------------------
// Tiled bf16 WMMA GEMM:  C[M,N] = A[M,K] @ Bt[N,K]^T  (+ epilogue)
// BM=BN=128, BK=32, 256 threads = 8 waves, wave -> 64x32 (4x2 WMMA tiles).
// Double-buffered LDS, tiles staged with global_load_async_to_lds_b128.
// ---------------------------------------------------------------------------
enum { EP_BF16 = 0, EP_BIAS_GELU_BF16 = 1, EP_RES_F32 = 2, EP_BIAS_RES_F32 = 3 };

#define GBM 128
#define GBN 128
#define GBK 32
#define GLDS 40  // padded stride: 80B rows (20 dwords, conflict-free, 16B align)

template <int EPI>
__global__ __launch_bounds__(256) void gemm_bf16_k(
    const bf16_t* __restrict__ A, const bf16_t* __restrict__ Bt, int M, int N,
    int K, const float* __restrict__ bias, const float* __restrict__ resid,
    void* __restrict__ Cout) {
  __shared__ __align__(16) bf16_t lA[2][GBM * GLDS];
  __shared__ __align__(16) bf16_t lB[2][GBN * GLDS];
  (void)M;

  int tid   = threadIdx.x;
  int wave  = tid >> 5;
  int lane  = tid & 31;
  int tileN = blockIdx.x * GBN;
  int tileM = blockIdx.y * GBM;

  int wm = (wave >> 2) * 64;
  int wn = (wave & 3) * 32;

  v8f acc[4][2];
#pragma unroll
  for (int i = 0; i < 4; ++i)
#pragma unroll
    for (int j = 0; j < 2; ++j)
#pragma unroll
      for (int v = 0; v < 8; ++v) acc[i][j][v] = 0.f;

  // staging map: thread -> one row, 16-element half (2x b128 per tile matrix)
  int sr = tid >> 1;             // 0..127
  int sc2 = (tid & 1) * 16;      // 0 / 16
  const bf16_t* gA = A + (size_t)(tileM + sr) * K + sc2;
  const bf16_t* gB = Bt + (size_t)(tileN + sr) * K + sc2;
  int lo = sr * GLDS + sc2;

  // preload k-tile 0 into buffer 0 (4 async ops per thread per stage)
  async_b128(gA, &lA[0][lo]);
  async_b128(gA + 8, &lA[0][lo + 8]);
  async_b128(gB, &lB[0][lo]);
  async_b128(gB + 8, &lB[0][lo + 8]);

  int nk = K / GBK;
  for (int k = 0; k < nk; ++k) {
    int cur = k & 1;
    if (k + 1 < nk) {
      int nxt = cur ^ 1;
      int k0 = (k + 1) * GBK;
      async_b128(gA + k0, &lA[nxt][lo]);
      async_b128(gA + k0 + 8, &lA[nxt][lo + 8]);
      async_b128(gB + k0, &lB[nxt][lo]);
      async_b128(gB + k0 + 8, &lB[nxt][lo + 8]);
      wait_async4();  // previous stage (4 ops) landed, next still in flight
    } else {
      wait_async0();
    }
    __syncthreads();

    v16bf af[4], bfr[2];
#pragma unroll
    for (int i = 0; i < 4; ++i) af[i] = load_frag<GLDS>(lA[cur], wm + i * 16, 0);
#pragma unroll
    for (int j = 0; j < 2; ++j) bfr[j] = load_frag<GLDS>(lB[cur], wn + j * 16, 0);
#pragma unroll
    for (int i = 0; i < 4; ++i)
#pragma unroll
      for (int j = 0; j < 2; ++j) acc[i][j] = wmma_bf16(af[i], bfr[j], acc[i][j]);
    __syncthreads();  // all waves done reading buf before it is re-staged
  }

  // epilogue (C layout: col = lane&15, row = v + 8*(lane>=16))
  int colL = lane & 15;
  int rowL = (lane >> 4) * 8;
#pragma unroll
  for (int i = 0; i < 4; ++i)
#pragma unroll
    for (int j = 0; j < 2; ++j) {
      int row0 = tileM + wm + i * 16 + rowL;
      int col  = tileN + wn + j * 16 + colL;
#pragma unroll
      for (int v = 0; v < 8; ++v) {
        size_t idx = (size_t)(row0 + v) * N + col;
        float x = acc[i][j][v];
        if (EPI == EP_BF16) {
          ((bf16_t*)Cout)[idx] = f2bf(x);
        } else if (EPI == EP_BIAS_GELU_BF16) {
          x += bias[col];
          x = 0.5f * x * (1.f + erff(x * 0.70710678118f));
          ((bf16_t*)Cout)[idx] = f2bf(x);
        } else if (EPI == EP_RES_F32) {
          ((float*)Cout)[idx] = x + resid[idx];
        } else {  // EP_BIAS_RES_F32
          ((float*)Cout)[idx] = x + bias[col] + resid[idx];
        }
      }
    }
}

// ---------------------------------------------------------------------------
// Flash attention: one (b,h) per blockIdx.y, BR=128 rows (8 waves x 16),
// BC=64 keys per step, HEAD_DIM=64, scale=1/8.
// qkv: [b*n][3072] bf16 (Q | K | V each [h*64+d]).  Q/K staged async.
// ---------------------------------------------------------------------------
#define ABR 128
#define ABC 64
#define ALD 72  // 144B rows (36 dwords, conflict-free, 16B aligned)

__global__ __launch_bounds__(256) void attention_k(
    const bf16_t* __restrict__ qkv, bf16_t* __restrict__ out, int n) {
  __shared__ __align__(16) bf16_t lQ[ABR * ALD];
  __shared__ __align__(16) bf16_t lK[ABC * ALD];     // [c][d]
  __shared__ __align__(16) bf16_t lVt[64 * ALD];     // [d][c]
  __shared__ __align__(16) bf16_t lP[8 * 16 * ALD];  // per-wave P blocks

  const int QS = 3072;
  int bh = blockIdx.y;
  int b  = bh >> 4;
  int h  = bh & 15;
  int q0 = blockIdx.x * ABR;
  int tid = threadIdx.x, lane = tid & 31, wave = tid >> 5;

  const bf16_t* Qg = qkv + (size_t)(b * n) * QS + h * 64;
  const bf16_t* Kg = Qg + 1024;
  const bf16_t* Vg = Qg + 2048;

  {  // stage Q tile 128x64 asynchronously
    int r = tid >> 1, c = (tid & 1) * 32;
    const bf16_t* gp = Qg + (size_t)(q0 + r) * QS + c;
    bf16_t* lp = &lQ[r * ALD + c];
    async_b128(gp, lp);
    async_b128(gp + 8, lp + 8);
    async_b128(gp + 16, lp + 16);
    async_b128(gp + 24, lp + 24);
  }

  float m_run[8], l_run[8];
  v8f accO[4];
#pragma unroll
  for (int v = 0; v < 8; ++v) { m_run[v] = -1e30f; l_run[v] = 0.f; }
#pragma unroll
  for (int t = 0; t < 4; ++t)
#pragma unroll
    for (int v = 0; v < 8; ++v) accO[t][v] = 0.f;

  int rowL = (lane >> 4) * 8;
  int colL = lane & 15;
  bf16_t* Pw = &lP[wave * 16 * ALD];

  for (int kt = 0; kt < n; kt += ABC) {
    __syncthreads();  // all waves done with previous K/V/P reads
    {  // K tile 64x64: async straight copy
      int r = tid >> 2, c = (tid & 3) * 16;
      const bf16_t* gp = Kg + (size_t)(kt + r) * QS + c;
      bf16_t* lp = &lK[r * ALD + c];
      async_b128(gp, lp);
      async_b128(gp + 8, lp + 8);
    }
    {  // V tile transposed -> [d][c] (VGPR path, needs scatter)
      int r = tid >> 2, c = (tid & 3) * 16;
      const bf16_t* gp = Vg + (size_t)(kt + r) * QS + c;
      v8bf x0 = *(const v8bf*)gp;
      v8bf x1 = *(const v8bf*)(gp + 8);
#pragma unroll
      for (int j = 0; j < 8; ++j) {
        lVt[(c + j) * ALD + r]     = x0[j];
        lVt[(c + 8 + j) * ALD + r] = x1[j];
      }
    }
    wait_async0();  // Q (first iter) + K landed
    __syncthreads();

    // S = scale * Q K^T
    v8f accS[4];
#pragma unroll
    for (int j = 0; j < 4; ++j)
#pragma unroll
      for (int v = 0; v < 8; ++v) accS[j][v] = 0.f;
#pragma unroll
    for (int kh = 0; kh < 64; kh += 32) {
      v16bf a = load_frag<ALD>(lQ, wave * 16, kh);
#pragma unroll
      for (int j = 0; j < 4; ++j) {
        v16bf bb = load_frag<ALD>(lK, j * 16, kh);
        accS[j] = wmma_bf16(a, bb, accS[j]);
      }
    }
#pragma unroll
    for (int j = 0; j < 4; ++j)
#pragma unroll
      for (int v = 0; v < 8; ++v) accS[j][v] *= 0.125f;

    // online softmax (row = rowL+v lives within one 16-lane half)
#pragma unroll
    for (int v = 0; v < 8; ++v) {
      float mx = -1e30f;
#pragma unroll
      for (int j = 0; j < 4; ++j) mx = fmaxf(mx, accS[j][v]);
#pragma unroll
      for (int msk = 1; msk < 16; msk <<= 1) mx = fmaxf(mx, __shfl_xor(mx, msk, 32));
      float mnew = fmaxf(m_run[v], mx);
      float corr = __expf(m_run[v] - mnew);
      m_run[v]   = mnew;
      float rs = 0.f;
#pragma unroll
      for (int j = 0; j < 4; ++j) {
        float p = __expf(accS[j][v] - mnew);
        accS[j][v] = p;
        rs += p;
      }
#pragma unroll
      for (int msk = 1; msk < 16; msk <<= 1) rs += __shfl_xor(rs, msk, 32);
      l_run[v] = l_run[v] * corr + rs;
#pragma unroll
      for (int t = 0; t < 4; ++t) accO[t][v] *= corr;
    }

    // P (C layout) -> wave-private LDS in A layout
#pragma unroll
    for (int j = 0; j < 4; ++j)
#pragma unroll
      for (int v = 0; v < 8; ++v)
        Pw[(rowL + v) * ALD + j * 16 + colL] = f2bf(accS[j][v]);
    __syncthreads();

    // O += P V
#pragma unroll
    for (int kh = 0; kh < 64; kh += 32) {
      v16bf a = load_frag<ALD>(Pw, 0, kh);
#pragma unroll
      for (int t = 0; t < 4; ++t) {
        v16bf bb = load_frag<ALD>(lVt, t * 16, kh);
        accO[t] = wmma_bf16(a, bb, accO[t]);
      }
    }
  }

  // finalize: O /= l, write bf16 to [b*n][1024]
#pragma unroll
  for (int t = 0; t < 4; ++t)
#pragma unroll
    for (int v = 0; v < 8; ++v) {
      float o = accO[t][v] / l_run[v];
      int row = b * n + q0 + wave * 16 + rowL + v;
      int col = h * 64 + t * 16 + colL;
      out[(size_t)row * 1024 + col] = f2bf(o);
    }
}

// ---------------------------------------------------------------------------
// Host orchestration
// ---------------------------------------------------------------------------
extern "C" void kernel_launch(void* const* d_in, const int* in_sizes, int n_in,
                              void* d_out, int out_size, void* d_ws,
                              size_t ws_size, hipStream_t stream) {
  (void)in_sizes; (void)n_in; (void)out_size; (void)ws_size;
  const float* x_in  = (const float*)d_in[0];
  const float* Wqkv  = (const float*)d_in[1];
  const float* Wout  = (const float*)d_in[2];
  const float* ln1s  = (const float*)d_in[3];
  const float* ln1b  = (const float*)d_in[4];
  const float* W1    = (const float*)d_in[5];
  const float* b1    = (const float*)d_in[6];
  const float* W2    = (const float*)d_in[7];
  const float* b2    = (const float*)d_in[8];
  const float* ln2s  = (const float*)d_in[9];
  const float* ln2b  = (const float*)d_in[10];
  const float* lnfs  = (const float*)d_in[11];
  const float* lnfb  = (const float*)d_in[12];

  const int BT = 4096;  // b * n tokens
  const int D = 1024, F = 4096, NL = 4, SEQ = 2048;

  char* ws = (char*)d_ws;
  size_t off = 0;
  auto alloc = [&](size_t bytes) -> void* {
    void* p = ws + off;
    off += (bytes + 255) & ~(size_t)255;
    return p;
  };
  // weights stored TRANSPOSED: [L][N][K] bf16
  bf16_t* wqkv_b = (bf16_t*)alloc((size_t)NL * 3 * D * D * 2);
  bf16_t* wout_b = (bf16_t*)alloc((size_t)NL * D * D * 2);
  bf16_t* w1_b   = (bf16_t*)alloc((size_t)NL * F * D * 2);
  bf16_t* w2_b   = (bf16_t*)alloc((size_t)NL * D * F * 2);
  float*  xres   = (float*)alloc((size_t)BT * D * 4);
  bf16_t* hbuf   = (bf16_t*)alloc((size_t)BT * D * 2);
  bf16_t* qkv    = (bf16_t*)alloc((size_t)BT * 3 * D * 2);
  bf16_t* attnb  = (bf16_t*)alloc((size_t)BT * D * 2);
  bf16_t* ffh    = (bf16_t*)alloc((size_t)BT * F * 2);

  dim3 blk(256);
  for (int L = 0; L < NL; ++L) {  // convert+transpose each layer's weights
    convert_transpose_k<<<dim3(3 * D / 32, D / 32), blk, 0, stream>>>(
        Wqkv + (size_t)L * D * 3 * D, wqkv_b + (size_t)L * 3 * D * D, D, 3 * D);
    convert_transpose_k<<<dim3(D / 32, D / 32), blk, 0, stream>>>(
        Wout + (size_t)L * D * D, wout_b + (size_t)L * D * D, D, D);
    convert_transpose_k<<<dim3(F / 32, D / 32), blk, 0, stream>>>(
        W1 + (size_t)L * D * F, w1_b + (size_t)L * F * D, D, F);
    convert_transpose_k<<<dim3(D / 32, F / 32), blk, 0, stream>>>(
        W2 + (size_t)L * F * D, w2_b + (size_t)L * D * F, F, D);
  }
  hipMemcpyAsync(xres, x_in, (size_t)BT * D * sizeof(float),
                 hipMemcpyDeviceToDevice, stream);

  for (int L = 0; L < NL; ++L) {
    // attention block
    layernorm_k<false><<<BT, blk, 0, stream>>>(xres, ln1s + L * D, ln1b + L * D,
                                               hbuf, D);
    gemm_bf16_k<EP_BF16><<<dim3(3 * D / GBN, BT / GBM), blk, 0, stream>>>(
        hbuf, wqkv_b + (size_t)L * 3 * D * D, BT, 3 * D, D, nullptr, nullptr,
        qkv);
    attention_k<<<dim3(SEQ / ABR, 32), blk, 0, stream>>>(qkv, attnb, SEQ);
    gemm_bf16_k<EP_RES_F32><<<dim3(D / GBN, BT / GBM), blk, 0, stream>>>(
        attnb, wout_b + (size_t)L * D * D, BT, D, D, nullptr, xres, xres);
    // feedforward block
    layernorm_k<false><<<BT, blk, 0, stream>>>(xres, ln2s + L * D, ln2b + L * D,
                                               hbuf, D);
    gemm_bf16_k<EP_BIAS_GELU_BF16><<<dim3(F / GBN, BT / GBM), blk, 0, stream>>>(
        hbuf, w1_b + (size_t)L * F * D, BT, F, D, b1 + L * F, nullptr, ffh);
    gemm_bf16_k<EP_BIAS_RES_F32><<<dim3(D / GBN, BT / GBM), blk, 0, stream>>>(
        ffh, w2_b + (size_t)L * D * F, BT, D, F, b2 + L * D, xres, xres);
  }
  layernorm_k<true><<<BT, blk, 0, stream>>>(xres, lnfs, lnfb, d_out, D);
}
